// MambaBlock_69140383531158
// MI455X (gfx1250) — matrix operand order repaired and checked by reading, not compiled, not run on hardware
//
#include <hip/hip_runtime.h>
#include <hip/hip_bf16.h>
#include <math.h>

#define D_MODEL 256
#define N_LAYER 2
#define D_STATE 16
#define D_CONV  4
#define D_INNER 1024
#define BATCH   2
#define SEQLEN  1024
#define TOK     (BATCH * SEQLEN)
#define SCHUNK  64   // scan timesteps staged per TDM tile

typedef __attribute__((ext_vector_type(16))) _Float16 v16h;
typedef __attribute__((ext_vector_type(8)))  float    v8f;
typedef __attribute__((ext_vector_type(4)))  unsigned int v4ui;
typedef __attribute__((ext_vector_type(8)))  int      v8si;
typedef __attribute__((ext_vector_type(4)))  int      v4si;

#if defined(__has_builtin)
#if __has_builtin(__builtin_amdgcn_tensor_load_to_lds)
#define HAVE_TDM 1
#endif
#endif
#ifndef HAVE_TDM
#define HAVE_TDM 0
#endif

// ---------------- weight f32 -> f16 conversion ----------------
__global__ __launch_bounds__(256) void k_cvt(const float* __restrict__ s,
                                             _Float16* __restrict__ d, int n) {
  int i = blockIdx.x * 256 + threadIdx.x;
  if (i < n) d[i] = (_Float16)s[i];
}

// ---------------- build per-direction residual stream (dir1 flipped) ----------------
__global__ __launch_bounds__(256) void k_prep(const float* __restrict__ x,
                                              float* __restrict__ resi) {
  int i = blockIdx.x * 256 + threadIdx.x;        // 2*2*1024*256 = 1<<20
  int c = i & 255, l = (i >> 8) & 1023, b = (i >> 18) & 1, d = i >> 19;
  int ls = d ? (SEQLEN - 1 - l) : l;
  resi[i] = x[((long)b * SEQLEN + ls) * D_MODEL + c];
}

// ---------------- RMSNorm: one wave per 256-wide row ----------------
__global__ __launch_bounds__(256) void k_rmsnorm(const float* __restrict__ resi,
                                                 const float* __restrict__ nw,
                                                 float* __restrict__ xn) {
  int wave = threadIdx.x >> 5, lane = threadIdx.x & 31;
  int row = blockIdx.x * 8 + wave;               // 4096 rows = [d][b][l]
  int d = row >> 11;
  const float* r = resi + (long)row * D_MODEL;
  float v[8]; float ss = 0.f;
#pragma unroll
  for (int i = 0; i < 8; i++) { v[i] = r[lane + 32 * i]; ss += v[i] * v[i]; }
#pragma unroll
  for (int m = 16; m >= 1; m >>= 1) ss += __shfl_xor(ss, m, 32);
  float sc = rsqrtf(ss * (1.f / D_MODEL) + 1e-5f);
  const float* w = nw + (long)d * (N_LAYER * D_MODEL);
  float* o = xn + (long)row * D_MODEL;
#pragma unroll
  for (int i = 0; i < 8; i++) o[lane + 32 * i] = v[i] * sc * w[lane + 32 * i];
}

// ---------------- WMMA GEMM: C[m,n] = sum_k A[m,k] * W[n,k] ----------------
// EPI: 0 = store, 1 = softplus(acc + bias[n]), 2 = C += acc (residual accumulate)
template <bool GK, int EPI>
__global__ __launch_bounds__(256) void k_gemm(
    const float* __restrict__ A, long dA, int lda,
    const _Float16* __restrict__ W, long dW,
    float* __restrict__ C, long dC, int ldc,
    const float* __restrict__ bias, long dBias,
    int M, int N, int K) {
  int d = blockIdx.y;
  const float*    Ad = A + (long)d * dA;
  const _Float16* Wd = W + (long)d * dW;
  float*          Cd = C + (long)d * dC;
  int wave = threadIdx.x >> 5, lane = threadIdx.x & 31;
  int nt_n = N >> 4;
  int tile = blockIdx.x * 8 + wave;
  if (tile >= (M >> 4) * nt_n) return;
  int m0 = (tile / nt_n) << 4, n0 = (tile % nt_n) << 4;
  int r = lane & 15, grp = lane >> 4;
  const float*    arow = Ad + (long)(m0 + r) * lda;
  const _Float16* brow = Wd + (long)(n0 + r) * K;
  v8f acc = {};
  for (int k0 = 0; k0 < K; k0 += 32) {
    v16h af, bf;
    if (!GK) {
#pragma unroll
      for (int i = 0; i < 8; i++) af[i]     = (_Float16)arow[k0 + 16 * grp + i];
#pragma unroll
      for (int i = 0; i < 8; i++) af[8 + i] = (_Float16)arow[k0 + 16 * grp + 8 + i];
      bf = *(const v16h*)(brow + k0 + 16 * grp);   // 32B-aligned contiguous halves
    } else {
#pragma unroll
      for (int i = 0; i < 8; i++) { int k = k0 + 16 * grp + i;     af[i]     = k < K ? (_Float16)arow[k] : (_Float16)0.f; }
#pragma unroll
      for (int i = 0; i < 8; i++) { int k = k0 + 16 * grp + 8 + i; af[8 + i] = k < K ? (_Float16)arow[k] : (_Float16)0.f; }
#pragma unroll
      for (int i = 0; i < 16; i++) { int k = k0 + 16 * grp + i;    bf[i]     = k < K ? brow[k] : (_Float16)0.f; }
    }
    acc = __builtin_amdgcn_wmma_f32_16x16x32_f16(false, af, false, bf,
                                                 (short)0, acc, false, false);
  }
  int col = n0 + r;
#pragma unroll
  for (int vv = 0; vv < 8; ++vv) {
    long idx = (long)(m0 + vv + 8 * grp) * ldc + col;
    float val = acc[vv];
    if (EPI == 1) {
      float t = val + bias[(long)d * dBias + col];
      Cd[idx] = t > 20.f ? t : log1pf(__expf(t));
    } else if (EPI == 2) {
      Cd[idx] += val;
    } else {
      Cd[idx] = val;
    }
  }
}

// ---------------- causal depthwise conv (width 4) + SiLU ----------------
__global__ __launch_bounds__(256) void k_conv(const float* __restrict__ xz,
                                              const float* __restrict__ cw,
                                              const float* __restrict__ cb,
                                              float* __restrict__ xc) {
  long i = (long)blockIdx.x * 256 + threadIdx.x; // 2*2*1024*1024 = 1<<22
  int c = i & 1023, l = (int)((i >> 10) & 1023), db = (int)(i >> 20); // db = d*2+b
  int d = db >> 1;
  const float* base = xz + (long)db * SEQLEN * 2048;
  float acc = cb[(long)d * (N_LAYER * D_INNER) + c];
  const float* w = cw + (long)d * (N_LAYER * D_INNER * D_CONV) + (long)c * D_CONV;
#pragma unroll
  for (int k = 0; k < D_CONV; k++) {
    int ll = l - (D_CONV - 1) + k;
    if (ll >= 0) acc += base[(long)ll * 2048 + c] * w[k];
  }
  xc[i] = acc / (1.f + __expf(-acc));            // silu
}

// ---------------- selective scan: 1 thread per (channel, state) ----------------
// B/C rows (dbl) are shared by all 16 channels of a block -> staged into LDS by
// the Tensor Data Mover, double-buffered, issued by wave 0, TENSORcnt-tracked.
__global__ __launch_bounds__(256) void k_scan(
    const float* __restrict__ xc, const float* __restrict__ dt,
    const float* __restrict__ dbl, const float* __restrict__ xz,
    const float* __restrict__ Al, const float* __restrict__ Dp,
    float* __restrict__ ymul) {
  int d = blockIdx.y;
  int b = blockIdx.x >> 6;
  int c = ((blockIdx.x & 63) << 4) + (threadIdx.x >> 4);
  int n = threadIdx.x & 15;
  float Aval = -__expf(Al[(long)d * (N_LAYER * D_INNER * D_STATE) + (long)c * D_STATE + n]);
  float Dc   = Dp[(long)d * (N_LAYER * D_INNER) + c];
  long tokbase = ((long)d * BATCH + b) * SEQLEN;
  const float* xcb  = xc  + tokbase * D_INNER;
  const float* dtp  = dt  + tokbase * D_INNER;
  const float* dblb = dbl + tokbase * 48;
  const float* xzb  = xz  + tokbase * 2048;
  float* yb = ymul + tokbase * D_INNER;

#if HAVE_TDM
  __shared__ float sB[2][SCHUNK * 48];
  bool issuer = (threadIdx.x < 32);           // wave 0 drives the TDM
  auto issue = [&](int chunk, int buf) {
    unsigned long long ga = (unsigned long long)(dblb + (long)chunk * SCHUNK * 48);
    unsigned lds = (unsigned)(unsigned long long)&sB[buf][0];
    v4ui g0;
    g0[0] = 1u;                                         // count=1, user descriptor
    g0[1] = lds;                                        // lds_addr [63:32]
    g0[2] = (unsigned)ga;                               // global_addr[31:0]
    g0[3] = (unsigned)((ga >> 32) & 0x01FFFFFFu) | (2u << 30); // addr[56:32] | type=2
    v8si g1;
    g1[0] = (int)(2u << 16);                            // data_size = 4B
    g1[1] = (int)(48u << 16);                           // tensor_dim0 = 48 (lo16)
    g1[2] = 0;                                          // dim0 hi / dim1 lo
    g1[3] = 16 | (int)(48u << 16);                      // tensor_dim1=1<<20 (hi16) | tile_dim0=48
    g1[4] = SCHUNK;                                     // tile_dim1 = 64, tile_dim2 = 0
    g1[5] = 48;                                         // tensor_dim0_stride = 48
    g1[6] = 0; g1[7] = 0;
    v4si z4 = {0, 0, 0, 0};
#if defined(__clang_major__) && (__clang_major__ >= 23)
    v8si z8 = {0, 0, 0, 0, 0, 0, 0, 0};
    __builtin_amdgcn_tensor_load_to_lds(g0, g1, z4, z4, z8, 0);
#else
    __builtin_amdgcn_tensor_load_to_lds(g0, g1, z4, z4, 0);
#endif
  };
  if (issuer) { issue(0, 0); __builtin_amdgcn_s_wait_tensorcnt(0); }
  __syncthreads();
#endif

  float h = 0.f;
  for (int chunk = 0; chunk < SEQLEN / SCHUNK; ++chunk) {
#if HAVE_TDM
    if (issuer && chunk + 1 < SEQLEN / SCHUNK) issue(chunk + 1, (chunk + 1) & 1);
    const float* sb = sB[chunk & 1];
#endif
    for (int t = 0; t < SCHUNK; ++t) {
      int l = chunk * SCHUNK + t;
      float u   = xcb[(long)l * D_INNER + c];
      float dtv = dtp[(long)l * D_INNER + c];
#if HAVE_TDM
      float Bv = sb[t * 48 + 16 + n];
      float Cv = sb[t * 48 + 32 + n];
#else
      float Bv = dblb[(long)l * 48 + 16 + n];
      float Cv = dblb[(long)l * 48 + 32 + n];
#endif
      h = __expf(dtv * Aval) * h + dtv * u * Bv;
      float p = h * Cv;
#pragma unroll
      for (int m = 8; m >= 1; m >>= 1) p += __shfl_xor(p, m, 32); // 16-lane group sum
      if (n == 0) {
        float z = xzb[(long)l * 2048 + 1024 + c];
        yb[(long)l * D_INNER + c] = (p + u * Dc) * (z / (1.f + __expf(-z)));
      }
      if (l + 16 < SEQLEN) {
        __builtin_prefetch(&xcb[(long)(l + 16) * D_INNER + c], 0, 1);
        __builtin_prefetch(&dtp[(long)(l + 16) * D_INNER + c], 0, 1);
      }
    }
#if HAVE_TDM
    if (issuer && chunk + 1 < SEQLEN / SCHUNK) __builtin_amdgcn_s_wait_tensorcnt(0);
    __syncthreads();
#endif
  }
}

// ---------------- final writeout (unflip dir 1, concat channels) ----------------
__global__ __launch_bounds__(256) void k_out(const float* __restrict__ resi,
                                             float* __restrict__ out) {
  int i = blockIdx.x * 256 + threadIdx.x;        // 1<<20
  int c = i & 255, l = (i >> 8) & 1023, b = (i >> 18) & 1, d = i >> 19;
  int lo = d ? (SEQLEN - 1 - l) : l;
  out[((long)b * SEQLEN + lo) * 512 + d * 256 + c] = resi[i];
}

extern "C" void kernel_launch(void* const* d_in, const int* in_sizes, int n_in,
                              void* d_out, int out_size, void* d_ws, size_t ws_size,
                              hipStream_t stream) {
  const float* x      = (const float*)d_in[0];
  const float* norm_w = (const float*)d_in[1];
  const float* in_w   = (const float*)d_in[2];
  const float* conv_w = (const float*)d_in[3];
  const float* conv_b = (const float*)d_in[4];
  const float* xp_w   = (const float*)d_in[5];
  const float* dt_w   = (const float*)d_in[6];
  const float* dt_b   = (const float*)d_in[7];
  const float* A_log  = (const float*)d_in[8];
  const float* Dp     = (const float*)d_in[9];
  const float* out_w  = (const float*)d_in[10];
  float* out = (float*)d_out;

  char* p = (char*)d_ws;
  auto take = [&](size_t bytes) -> void* {
    void* r = (void*)p; p += (bytes + 255) & ~(size_t)255; return r;
  };
  _Float16* h_in = (_Float16*)take(sizeof(_Float16) * 2 * 2 * 2048 * 256);
  _Float16* h_xp = (_Float16*)take(sizeof(_Float16) * 2 * 2 * 48 * 1024);
  _Float16* h_dt = (_Float16*)take(sizeof(_Float16) * 2 * 2 * 1024 * 16);
  _Float16* h_ow = (_Float16*)take(sizeof(_Float16) * 2 * 2 * 256 * 1024);
  float* resi = (float*)take(sizeof(float) * 2L * TOK * 256);
  float* xn   = (float*)take(sizeof(float) * 2L * TOK * 256);
  float* xz   = (float*)take(sizeof(float) * 2L * TOK * 2048);
  float* xc   = (float*)take(sizeof(float) * 2L * TOK * 1024);
  float* dbl  = (float*)take(sizeof(float) * 2L * TOK * 48);
  float* dtb  = (float*)take(sizeof(float) * 2L * TOK * 1024);
  float* ym   = (float*)take(sizeof(float) * 2L * TOK * 1024);

  auto cdiv = [](long a, long b) { return (int)((a + b - 1) / b); };

  k_cvt<<<cdiv(2L * 2 * 2048 * 256, 256), 256, 0, stream>>>(in_w, h_in, 2 * 2 * 2048 * 256);
  k_cvt<<<cdiv(2L * 2 * 48 * 1024, 256), 256, 0, stream>>>(xp_w, h_xp, 2 * 2 * 48 * 1024);
  k_cvt<<<cdiv(2L * 2 * 1024 * 16, 256), 256, 0, stream>>>(dt_w, h_dt, 2 * 2 * 1024 * 16);
  k_cvt<<<cdiv(2L * 2 * 256 * 1024, 256), 256, 0, stream>>>(out_w, h_ow, 2 * 2 * 256 * 1024);
  k_prep<<<cdiv(1 << 20, 256), 256, 0, stream>>>(x, resi);

  for (int i = 0; i < N_LAYER; i++) {
    k_rmsnorm<<<512, 256, 0, stream>>>(resi, norm_w + (long)i * D_MODEL, xn);

    { // in_proj: M=2048 N=2048 K=256
      int tiles = (TOK / 16) * (2048 / 16);
      dim3 g(cdiv(tiles, 8), 2);
      k_gemm<false, 0><<<g, 256, 0, stream>>>(
          xn, (long)TOK * 256, 256,
          h_in + (long)i * 2048 * 256, 2L * 2048 * 256,
          xz, (long)TOK * 2048, 2048, nullptr, 0, TOK, 2048, 256);
    }
    k_conv<<<cdiv(1L << 22, 256), 256, 0, stream>>>(
        xz, conv_w + (long)i * D_INNER * D_CONV, conv_b + (long)i * D_INNER, xc);
    { // x_proj: M=2048 N=48 K=1024
      int tiles = (TOK / 16) * (48 / 16);
      dim3 g(cdiv(tiles, 8), 2);
      k_gemm<false, 0><<<g, 256, 0, stream>>>(
          xc, (long)TOK * 1024, 1024,
          h_xp + (long)i * 48 * 1024, 2L * 48 * 1024,
          dbl, (long)TOK * 48, 48, nullptr, 0, TOK, 48, 1024);
    }
    { // dt_proj + softplus: M=2048 N=1024 K=16 (zero-padded to 32)
      int tiles = (TOK / 16) * (1024 / 16);
      dim3 g(cdiv(tiles, 8), 2);
      k_gemm<true, 1><<<g, 256, 0, stream>>>(
          dbl, (long)TOK * 48, 48,
          h_dt + (long)i * 1024 * 16, 2L * 1024 * 16,
          dtb, (long)TOK * 1024, 1024, dt_b + (long)i * 1024, 2L * 1024, TOK, 1024, 16);
    }
    { // selective scan (+ u*D, * silu(z)), TDM-staged B/C
      dim3 g(128, 2);
      k_scan<<<g, 256, 0, stream>>>(xc, dtb, dbl, xz,
                                    A_log + (long)i * 1024 * 16, Dp + (long)i * 1024, ym);
    }
    { // out_proj, accumulated into residual stream: resi += ym @ ow.T
      int tiles = (TOK / 16) * (256 / 16);
      dim3 g(cdiv(tiles, 8), 2);
      k_gemm<false, 2><<<g, 256, 0, stream>>>(
          ym, (long)TOK * 1024, 1024,
          h_ow + (long)i * 256 * 1024, 2L * 256 * 1024,
          resi, (long)TOK * 256, 256, nullptr, 0, TOK, 256, 1024);
    }
  }
  k_out<<<cdiv(1 << 20, 256), 256, 0, stream>>>(resi, out);
}
